// CharacterLSTM_88330297409632
// MI455X (gfx1250) — matrix operand order repaired and checked by reading, not compiled, not run on hardware
//
#include <hip/hip_runtime.h>

typedef __attribute__((ext_vector_type(16))) _Float16     v16h;
typedef __attribute__((ext_vector_type(8)))  _Float16     v8h;
typedef __attribute__((ext_vector_type(8)))  float        v8f;
typedef __attribute__((ext_vector_type(4)))  unsigned int u32x4;

// explicit global-address-space pointers: force global_load_* (LOADcnt only)
// instead of flat_load_* (LOADcnt + DScnt) for the hot-loop streams.
typedef const v16h  __attribute__((address_space(1)))* gfrag_t;
typedef const u32x4 __attribute__((address_space(1)))* gvec4_t;

namespace {
constexpr int kB = 64, kS = 256, kW = 24, kE = 128, kH = 256, kV = 256;
constexpr int kN = kB * kS;     // 16384 words
constexpr int kG = 4 * kH;      // 1024 gate columns [i|f|g|o]

constexpr int TN      = 32;             // rows of N per workgroup (2 row tiles)
constexpr int NWAVES  = 16;
constexpr int THREADS = NWAVES * 32;    // 512
constexpr int RT      = TN / 16;        // 2
constexpr int CPW     = 4;              // col tiles per wave: one per gate
constexpr int KTX     = kE / 32;        // 4 k-tiles for x GEMM
constexpr int KTH     = kH / 32;        // 8 k-tiles for h GEMM

constexpr size_t WIH_HALFS = (size_t)kG * kE;   // 131072
constexpr size_t WHH_HALFS = (size_t)kG * kH;   // 262144
constexpr size_t EMB_HALFS = (size_t)kV * kE;   // 32768
} // namespace

// ---------------------------------------------------------------------------
// Pack weights into WMMA B-fragment layout (f16), embedding -> f16, bias fuse.
// B-matrix (32x16, 16-bit) lane layout used here (contiguous K, not the ISA table order; A matches):
//   lane l holds column n = l&15; lanes 0-15 carry K=0..15, lanes 16-31 K=16..31,
//   element j of the 16-half fragment is K = (l>=16?16:0) + j.
// Fragment id f = colTile*KT + kTile; storage: [f][lane][j] contiguous halves.
// ---------------------------------------------------------------------------
__global__ void lstm_pack(const float* __restrict__ Wih,
                          const float* __restrict__ Whh,
                          const float* __restrict__ emb,
                          const float* __restrict__ bih,
                          const float* __restrict__ bhh,
                          _Float16* __restrict__ wih_p,
                          _Float16* __restrict__ whh_p,
                          _Float16* __restrict__ emb_h,
                          float* __restrict__ bias) {
  long gid = (long)blockIdx.x * blockDim.x + threadIdx.x;
  if (gid < (long)WIH_HALFS) {
    int idx  = (int)gid;
    int f    = idx >> 9;                 // fragment id (512 halves each)
    int lane = (idx >> 4) & 31;
    int j    = idx & 15;
    int c    = f / KTX, kk = f % KTX;
    int row  = c * 16 + (lane & 15);                       // gate column g
    int col  = kk * 32 + ((lane >= 16) ? 16 : 0) + j;      // k index in E
    wih_p[idx] = (_Float16)Wih[row * kE + col];
    return;
  }
  gid -= (long)WIH_HALFS;
  if (gid < (long)WHH_HALFS) {
    int idx  = (int)gid;
    int f    = idx >> 9;
    int lane = (idx >> 4) & 31;
    int j    = idx & 15;
    int c    = f / KTH, kk = f % KTH;
    int row  = c * 16 + (lane & 15);
    int col  = kk * 32 + ((lane >= 16) ? 16 : 0) + j;      // k index in H
    whh_p[idx] = (_Float16)Whh[row * kH + col];
    return;
  }
  gid -= (long)WHH_HALFS;
  if (gid < (long)EMB_HALFS) {
    emb_h[gid] = (_Float16)emb[gid];
    return;
  }
  gid -= (long)EMB_HALFS;
  if (gid < kG) bias[gid] = bih[gid] + bhh[gid];
}

// A-matrix (16x32, 16-bit) per-lane fragment from LDS row.
// Lane l is row m = l&15; K is contiguous per lane:
//   halves [0..7]  = K = (l>=16?16:0) + 0..7
//   halves [8..15] = K = (l>=16?16:0) + 8..15
__device__ __forceinline__ v16h load_afrag(const _Float16* __restrict__ row, int hi8) {
  v8h lo = *(const v8h*)(row + 2 * hi8);
  v8h hi = *(const v8h*)(row + 2 * hi8 + 8);
  return __builtin_shufflevector(lo, hi, 0, 1, 2, 3, 4, 5, 6, 7,
                                 8, 9, 10, 11, 12, 13, 14, 15);
}

__device__ __forceinline__ float sigmoidf_(float x) {
  return 1.f / (1.f + __expf(-x));
}
__device__ __forceinline__ float tanhf_(float x) {
  // 2*sigmoid(2x)-1 : saturates cleanly to +/-1 without NaN
  return 2.f / (1.f + __expf(-2.f * x)) - 1.f;
}

// ---------------------------------------------------------------------------
// Fused LSTM recurrence. One workgroup = 32 word rows, 16 waves.
// Wave w owns gate-column tiles {w, w+16, w+32, w+48} -> i/f/g/o for hidden
// units u in [16w, 16w+16), so the LSTM cell update is wave-local.
// Weights are re-streamed from global (L2-resident) every timestep; an opaque
// uniform offset stops LICM from hoisting 384 VGPRs of fragments out of the
// t-loop, while keeping the loads' global-addrspace provenance intact.
// ---------------------------------------------------------------------------
__global__ void __launch_bounds__(THREADS, 1)
lstm_fused(const int* __restrict__ chars,
           const _Float16* __restrict__ wih_p,
           const _Float16* __restrict__ whh_p,
           const _Float16* __restrict__ emb_h,
           const float* __restrict__ bias,
           float* __restrict__ out) {
  __shared__ __align__(16) _Float16 lds_x[TN * kE];   // 8 KB  f16 x tile
  __shared__ __align__(16) _Float16 lds_h[TN * kH];   // 16 KB f16 h state
  __shared__ int lds_flag[TN];                        // "t == len-1" per row

  const int tid  = threadIdx.x;
  const int lane = tid & 31;
  const int wave = tid >> 5;
  const int hi8  = (lane >= 16) ? 8 : 0;   // K-offset for A, M-offset for C/D
  const int nlo  = lane & 15;
  const int n0   = blockIdx.x * TN;

  // zero initial hidden state
  for (int i = tid; i < TN * kH; i += THREADS) lds_h[i] = (_Float16)0.f;

  // bias per owned gate tile (constant across the 8 acc VGPRs of a tile)
  float bini[CPW];
#pragma unroll
  for (int gi = 0; gi < CPW; ++gi) {
    int ct = wave + gi * NWAVES;
    bini[gi] = bias[ct * 16 + nlo];
  }

  float cst[RT][8];   // cell state for (m = rt*16+hi8+v, u = wave*16+nlo)
  float lh[RT][8];    // captured last hidden
#pragma unroll
  for (int rt = 0; rt < RT; ++rt)
#pragma unroll
    for (int v = 0; v < 8; ++v) { cst[rt][v] = 0.f; lh[rt][v] = 0.f; }

  const gfrag_t wfrag_base = (gfrag_t)wih_p;
  const gfrag_t hfrag_base = (gfrag_t)whh_p;
  unsigned toff = 0;   // always 0, but opaque to the optimizer

  for (int t = 0; t < kW; ++t) {
    // Opaque update: keeps the weight-fragment loads inside the t-loop
    // (served from L2) instead of hoisted + spilled. Zero-cost.
    asm volatile("" : "+s"(toff));
    gfrag_t bih_frag = wfrag_base + toff;
    gfrag_t bhh_frag = hfrag_base + toff;

    // ---- phase A: per-row flags + gather x tile (f16 embeddings) ----------
    if (tid < TN) {
      int id  = chars[(n0 + tid) * kW + t];
      int nxt = (t + 1 < kW) ? chars[(n0 + tid) * kW + t + 1] : 0;
      lds_flag[tid] = (id != 0) && (nxt == 0 || t == kW - 1);
    }
    {
      int r   = tid >> 4;        // 0..31
      int seg = tid & 15;        // 16-byte chunk (8 halves) of 256B row
      int id  = chars[(n0 + r) * kW + t];
      gvec4_t src = (gvec4_t)(emb_h + (size_t)id * kE) + seg;
      u32x4 val = *src;
      *(u32x4*)(&lds_x[r * kE + seg * 8]) = val;
    }
    __syncthreads();   // x, h visible to all waves

    // snapshot capture flags into registers (avoid race with next A phase)
    unsigned myfl = 0;
#pragma unroll
    for (int rt = 0; rt < RT; ++rt)
#pragma unroll
      for (int v = 0; v < 8; ++v)
        if (lds_flag[rt * 16 + hi8 + v]) myfl |= 1u << (rt * 8 + v);

    // ---- phase B: gates = bias + x*Wih^T + h*Whh^T via WMMA ---------------
    v8f acc[CPW][RT];
#pragma unroll
    for (int gi = 0; gi < CPW; ++gi)
#pragma unroll
      for (int rt = 0; rt < RT; ++rt) {
        float bv = bini[gi];
#pragma unroll
        for (int v = 0; v < 8; ++v) acc[gi][rt][v] = bv;
      }

#pragma unroll
    for (int kk = 0; kk < KTX; ++kk) {
      v16h a0 = load_afrag(&lds_x[nlo * kE + kk * 32], hi8);
      v16h a1 = load_afrag(&lds_x[(16 + nlo) * kE + kk * 32], hi8);
#pragma unroll
      for (int gi = 0; gi < CPW; ++gi) {
        int ct = wave + gi * NWAVES;
        v16h bf = bih_frag[(ct * KTX + kk) * 32 + lane];
        acc[gi][0] = __builtin_amdgcn_wmma_f32_16x16x32_f16(
            false, a0, false, bf, (short)0, acc[gi][0], false, false);
        acc[gi][1] = __builtin_amdgcn_wmma_f32_16x16x32_f16(
            false, a1, false, bf, (short)0, acc[gi][1], false, false);
      }
    }
#pragma unroll
    for (int kk = 0; kk < KTH; ++kk) {
      v16h a0 = load_afrag(&lds_h[nlo * kH + kk * 32], hi8);
      v16h a1 = load_afrag(&lds_h[(16 + nlo) * kH + kk * 32], hi8);
#pragma unroll
      for (int gi = 0; gi < CPW; ++gi) {
        int ct = wave + gi * NWAVES;
        v16h bf = bhh_frag[(ct * KTH + kk) * 32 + lane];
        acc[gi][0] = __builtin_amdgcn_wmma_f32_16x16x32_f16(
            false, a0, false, bf, (short)0, acc[gi][0], false, false);
        acc[gi][1] = __builtin_amdgcn_wmma_f32_16x16x32_f16(
            false, a1, false, bf, (short)0, acc[gi][1], false, false);
      }
    }
    __syncthreads();   // all waves done reading lds_h before we overwrite it

    // ---- phase C: wave-local LSTM cell update -----------------------------
    const int u = wave * 16 + nlo;
#pragma unroll
    for (int rt = 0; rt < RT; ++rt) {
#pragma unroll
      for (int v = 0; v < 8; ++v) {
        int m = rt * 16 + hi8 + v;
        float iv = sigmoidf_(acc[0][rt][v]);
        float fv = sigmoidf_(acc[1][rt][v]);
        float gv = tanhf_(acc[2][rt][v]);
        float ov = sigmoidf_(acc[3][rt][v]);
        float cc = fv * cst[rt][v] + iv * gv;
        cst[rt][v] = cc;
        float hv = ov * tanhf_(cc);
        lds_h[m * kH + u] = (_Float16)hv;
        if (myfl & (1u << (rt * 8 + v))) lh[rt][v] = hv;
      }
    }
  }

  // ---- write last_h (covers every output element; len==0 rows stay 0) ----
#pragma unroll
  for (int rt = 0; rt < RT; ++rt)
#pragma unroll
    for (int v = 0; v < 8; ++v) {
      int m = rt * 16 + hi8 + v;
      out[(size_t)(n0 + m) * kH + (wave * 16 + nlo)] = lh[rt][v];
    }
}

extern "C" void kernel_launch(void* const* d_in, const int* in_sizes, int n_in,
                              void* d_out, int out_size, void* d_ws, size_t ws_size,
                              hipStream_t stream) {
  (void)in_sizes; (void)n_in; (void)out_size; (void)ws_size;
  const int*   chars = (const int*)d_in[0];
  const float* emb   = (const float*)d_in[1];
  const float* Wih   = (const float*)d_in[2];
  const float* Whh   = (const float*)d_in[3];
  const float* bih   = (const float*)d_in[4];
  const float* bhh   = (const float*)d_in[5];

  _Float16* ws_h  = (_Float16*)d_ws;
  _Float16* wih_p = ws_h;
  _Float16* whh_p = ws_h + WIH_HALFS;
  _Float16* emb_h = ws_h + WIH_HALFS + WHH_HALFS;
  float*    bias  = (float*)(ws_h + WIH_HALFS + WHH_HALFS + EMB_HALFS);

  long total = (long)(WIH_HALFS + WHH_HALFS + EMB_HALFS) + kG;
  int  pblk  = 256;
  int  pgrid = (int)((total + pblk - 1) / pblk);
  lstm_pack<<<pgrid, pblk, 0, stream>>>(Wih, Whh, emb, bih, bhh,
                                        wih_p, whh_p, emb_h, bias);

  lstm_fused<<<kN / TN, THREADS, 0, stream>>>(chars, wih_p, whh_p, emb_h, bias,
                                              (float*)d_out);
}
